// QuantGRULayer_42039139893405
// MI455X (gfx1250) — compile-verified, matrix operand order checked by reading, not run on hardware
//
#include <hip/hip_runtime.h>
#include <hip/hip_bf16.h>

typedef __attribute__((ext_vector_type(16))) _Float16 v16h;
typedef __attribute__((ext_vector_type(8)))  _Float16 v8h;
typedef __attribute__((ext_vector_type(8)))  float    v8f;
typedef __attribute__((ext_vector_type(4)))  int      v4i;

#define H_   512
#define IN_  512
#define T_   512
#define B_   64
#define NG   1536        // 3*H : [r | c | n] gate columns
#define M_   (T_ * B_)   // 32768 rows of the input-projection GEMM
#define RBLOCKS 16       // workgroups in the persistent recurrent kernel
#define RTPB    64       // threads per recurrent block (2 waves -> 1 wave/SIMD pair)
#define HPAD   528       // padded LDS row stride (halves): 264 dwords == 8 mod 64
                         // -> optimal 2-way bank pattern for 32-lane x 16B frag reads

#if defined(__has_builtin)
#if __has_builtin(__builtin_amdgcn_global_load_async_to_lds_b128)
#define ASYNC_LDS 1
#endif
#endif

typedef __attribute__((address_space(1))) v4i gv4i;   // global 16B chunk
typedef __attribute__((address_space(3))) v4i lv4i;   // LDS    16B chunk

// ---------------------------------------------------------------- helpers

__device__ __forceinline__ float fast_sigmoid(float x) {
    return __builtin_amdgcn_rcpf(1.0f + __expf(-x));
}
__device__ __forceinline__ float fast_tanh(float x) {
    return 1.0f - 2.0f * __builtin_amdgcn_rcpf(__expf(2.0f * x) + 1.0f);
}
__device__ __forceinline__ float q_sigmoid(float x) {
    return rintf(fast_sigmoid(x) * 255.0f) * (1.0f / 255.0f);
}
__device__ __forceinline__ float q_tanh(float x) {
    return rintf(fast_tanh(x) * 127.0f) * (1.0f / 127.0f);
}

// A/B fragment for v_wmma_f32_16x16x32_f16 from a row-major f16 matrix [rows, ld].
// ISA 16-bit A layout: lanes 0-15 -> M=row0+lane, K = k0+{0..7, 16..23};
// lanes 16-31 -> same rows, K = k0+{8..15, 24..31}. B (stored transposed, [N,K])
// uses the identical pattern with N in place of M.
__device__ __forceinline__ v16h ld_frag16(const _Float16* __restrict__ base,
                                          int row0, int k0, int ld) {
    const int lane = threadIdx.x & 31;
    const _Float16* p = base + (size_t)(row0 + (lane & 15)) * ld
                             + k0 + ((lane >> 4) << 3);
    v16h f;
#pragma unroll
    for (int i = 0; i < 8; ++i) { f[i] = p[i]; f[8 + i] = p[16 + i]; }
    return f;
}

// Same fragment but sourced from an f32 matrix (convert inline).
__device__ __forceinline__ v16h ld_frag32(const float* __restrict__ base,
                                          int row0, int k0, int ld) {
    const int lane = threadIdx.x & 31;
    const float* p = base + (size_t)(row0 + (lane & 15)) * ld
                          + k0 + ((lane >> 4) << 3);
    v16h f;
#pragma unroll
    for (int i = 0; i < 8; ++i) {
        f[i]     = (_Float16)p[i];
        f[8 + i] = (_Float16)p[16 + i];
    }
    return f;
}

// 16-byte global -> LDS copy: async (ASYNCcnt-tracked) op when the toolchain
// has the builtin; plain load+ds_store otherwise.
__device__ __forceinline__ void cp16B(const _Float16* g, _Float16* l) {
#ifdef ASYNC_LDS
    __builtin_amdgcn_global_load_async_to_lds_b128((gv4i*)g, (lv4i*)l, 0, 0);
#else
    *(v8h*)l = *(const v8h*)g;
#endif
}
__device__ __forceinline__ void stage_join() {
#ifdef ASYNC_LDS
#if __has_builtin(__builtin_amdgcn_s_wait_asynccnt)
    __builtin_amdgcn_s_wait_asynccnt(0);
#endif
#endif
    __syncthreads();
}

// ------------------------------------------------- K1: per-tensor absmax scale

__global__ void absmax_scale_kernel(const float* __restrict__ w0, const float* __restrict__ w1,
                                    const float* __restrict__ w2, const float* __restrict__ w3,
                                    const float* __restrict__ w4, const float* __restrict__ w5,
                                    float* __restrict__ scales) {
    __shared__ float red[256];
    const float* w = (blockIdx.x == 0) ? w0 : (blockIdx.x == 1) ? w1 :
                     (blockIdx.x == 2) ? w2 : (blockIdx.x == 3) ? w3 :
                     (blockIdx.x == 4) ? w4 : w5;
    float m = 0.0f;
    for (int i = threadIdx.x; i < H_ * IN_; i += 256)
        m = fmaxf(m, fabsf(w[i]));
    red[threadIdx.x] = m;
    __syncthreads();
    for (int s = 128; s > 0; s >>= 1) {
        if (threadIdx.x < (unsigned)s)
            red[threadIdx.x] = fmaxf(red[threadIdx.x], red[threadIdx.x + s]);
        __syncthreads();
    }
    if (threadIdx.x == 0)
        scales[blockIdx.x] = fmaxf(red[0] * (1.0f / 127.0f), 1e-8f);
}

// --------------------------- K2: fake-quantize + transpose weights to f16 [N,K]

__global__ void quant_transpose_kernel(const float* __restrict__ wa,
                                       const float* __restrict__ wb,
                                       const float* __restrict__ wc,
                                       const float* __restrict__ scales, int sbase,
                                       _Float16* __restrict__ outT) {
    int idx = blockIdx.x * 256 + threadIdx.x;           // over NG*K
    if (idx >= NG * IN_) return;
    int n = idx / IN_;
    int k = idx - n * IN_;
    int which = n >> 9;                                 // 0,1,2 -> a,b,c block
    int nn = n & (H_ - 1);
    const float* w = (which == 0) ? wa : (which == 1) ? wb : wc;
    float s = scales[sbase + which];
    float q = rintf(fminf(fmaxf(w[(size_t)k * H_ + nn] / s, -127.0f), 127.0f)) * s;
    outT[(size_t)n * IN_ + k] = (_Float16)q;            // transposed: [n][k]
}

// ------------- K3: input projection  pre[T*B, 3H] = X @ [Wri|Wci|Wni] + bias

__global__ __launch_bounds__(256)
void pre_gemm_kernel(const float* __restrict__ x,        // [M_, IN_]
                     const _Float16* __restrict__ WihT,  // [NG, IN_]
                     const float* __restrict__ bias_r,
                     const float* __restrict__ bias_c,
                     const float* __restrict__ bias_ni,
                     _Float16* __restrict__ pre) {       // [M_, NG]
    const int lane = threadIdx.x & 31;
    const int wave = threadIdx.x >> 5;
    const int row0 = blockIdx.x * 128 + wave * 16;       // 8 waves x 16 rows
    const int col0 = blockIdx.y * 64;                    // 4 column tiles / wave

    v8f acc[4] = {v8f{}, v8f{}, v8f{}, v8f{}};
    for (int k0 = 0; k0 < IN_; k0 += 32) {
        v16h a = ld_frag32(x, row0, k0, IN_);
#pragma unroll
        for (int c = 0; c < 4; ++c) {
            v16h b = ld_frag16(WihT, col0 + c * 16, k0, IN_);
            acc[c] = __builtin_amdgcn_wmma_f32_16x16x32_f16(
                false, a, false, b, (short)0, acc[c], false, false);
        }
    }
#pragma unroll
    for (int c = 0; c < 4; ++c) {
        int n = col0 + c * 16 + (lane & 15);
        float bias = (n < H_)     ? bias_r[n]
                   : (n < 2 * H_) ? bias_c[n - H_]
                                  : bias_ni[n - 2 * H_];
#pragma unroll
        for (int r = 0; r < 8; ++r) {
            int m = row0 + r + ((lane >> 4) << 3);
            pre[(size_t)m * NG + n] = (_Float16)(acc[c][r] + bias);
        }
    }
}

// --------------------- monotonic-counter grid barrier (single RMW + spin)

__device__ __forceinline__ void grid_barrier(unsigned* cnt, unsigned target) {
    __syncthreads();
    if (threadIdx.x == 0) {
        __threadfence();
        __hip_atomic_fetch_add(cnt, 1u, __ATOMIC_ACQ_REL, __HIP_MEMORY_SCOPE_AGENT);
        while (__hip_atomic_load(cnt, __ATOMIC_ACQUIRE,
                                 __HIP_MEMORY_SCOPE_AGENT) < target) {
            __builtin_amdgcn_s_sleep(1);
        }
    }
    __syncthreads();
}

// ---- K4: persistent recurrence. 16 blocks x 2 waves = 32 waves; wave w owns
// gate-column group j (16 cols) and computes r/c/n tiles with identical lane
// mapping, so the quantized gate math is fully register-resident. WhhT slices
// are staged into LDS once; h is staged into LDS each step (async when
// available), so the hot loop is ds_load_b128 + v_wmma.

__global__ __launch_bounds__(RTPB)
void recurrent_kernel(const _Float16* __restrict__ pre,   // [T*B, NG]
                      const _Float16* __restrict__ WhhT,  // [NG, H_]
                      const float* __restrict__ state,    // [B, H]
                      const float* __restrict__ bias_nh,  // [H]
                      float* __restrict__ out,            // [T,B,H] ++ [B,H]
                      float* __restrict__ hf32,           // [2][B,H]
                      _Float16* __restrict__ hf16,        // [2][B,H]
                      unsigned* __restrict__ bar) {
    extern __shared__ _Float16 smem[];
    _Float16* hs  = smem;                 // [B_=64][HPAD]   staged h_t (f16)
    _Float16* whs = smem + B_ * HPAD;     // [96][HPAD]      2 waves x 3 gates x 16 rows

    const int lane  = threadIdx.x & 31;
    const int wlocal = threadIdx.x >> 5;                       // 0..1
    const int wid   = (blockIdx.x << 1) + wlocal;              // 0..31
    const int n0    = wid * 16;
    const int coln  = n0 + (lane & 15);
    unsigned epoch = 0;

    // ---- stage this block's WhhT rows into LDS (loop-invariant), and init h
    for (int idx = threadIdx.x; idx < 96 * (H_ / 8); idx += RTPB) {
        int lrow = idx >> 6;                       // 0..95
        int c8   = idx & 63;
        int w    = lrow / 48;
        int rem  = lrow - w * 48;
        int g    = rem >> 4;
        int r    = rem & 15;
        int grow = g * H_ + ((blockIdx.x << 1) + w) * 16 + r;
        cp16B(WhhT + (size_t)grow * H_ + c8 * 8, whs + (size_t)lrow * HPAD + c8 * 8);
    }
    for (int i = blockIdx.x * RTPB + threadIdx.x; i < B_ * H_; i += RBLOCKS * RTPB) {
        float v = state[i];
        hf32[i] = v;
        hf16[i] = (_Float16)v;
    }
    stage_join();
    grid_barrier(bar, RBLOCKS * ++epoch);

    const float bnh = bias_nh[coln];
    const int wbase = wlocal * 48;        // this wave's rows in whs

    for (int t = 0; t < T_; ++t) {
        const _Float16* hc16 = hf16 + (size_t)(t & 1) * (B_ * H_);
        const float*    hc32 = hf32 + (size_t)(t & 1) * (B_ * H_);
        _Float16*       hn16 = hf16 + (size_t)((t + 1) & 1) * (B_ * H_);
        float*          hn32 = hf32 + (size_t)((t + 1) & 1) * (B_ * H_);

        // ---- stage h_t into LDS (async global->LDS, one copy shared by block)
        for (int idx = threadIdx.x; idx < B_ * (H_ / 8); idx += RTPB) {
            int row = idx >> 6;
            int c8  = idx & 63;
            cp16B(hc16 + (size_t)row * H_ + c8 * 8, hs + (size_t)row * HPAD + c8 * 8);
        }
        stage_join();

        v8f accR[4] = {v8f{}, v8f{}, v8f{}, v8f{}};
        v8f accC[4] = {v8f{}, v8f{}, v8f{}, v8f{}};
        v8f accN[4] = {v8f{}, v8f{}, v8f{}, v8f{}};

        for (int k0 = 0; k0 < H_; k0 += 32) {
            v16h bR = ld_frag16(whs, wbase,      k0, HPAD);
            v16h bC = ld_frag16(whs, wbase + 16, k0, HPAD);
            v16h bN = ld_frag16(whs, wbase + 32, k0, HPAD);
#pragma unroll
            for (int mt = 0; mt < 4; ++mt) {
                v16h a = ld_frag16(hs, mt * 16, k0, HPAD);
                accR[mt] = __builtin_amdgcn_wmma_f32_16x16x32_f16(
                    false, a, false, bR, (short)0, accR[mt], false, false);
                accC[mt] = __builtin_amdgcn_wmma_f32_16x16x32_f16(
                    false, a, false, bC, (short)0, accC[mt], false, false);
                accN[mt] = __builtin_amdgcn_wmma_f32_16x16x32_f16(
                    false, a, false, bN, (short)0, accN[mt], false, false);
            }
        }

        // fused quantized gate math + h update (lane holds matching r/c/n cols)
#pragma unroll
        for (int mt = 0; mt < 4; ++mt) {
#pragma unroll
            for (int r = 0; r < 8; ++r) {
                int m = mt * 16 + r + ((lane >> 4) << 3);
                const _Float16* prow = pre + (size_t)(t * B_ + m) * NG;
                float gr  = accR[mt][r] + (float)prow[coln];           // +bias_r
                float gc  = accC[mt][r] + (float)prow[H_ + coln];      // +bias_c
                float gni = (float)prow[2 * H_ + coln];                // +bias_ni
                float rg  = q_sigmoid(gr);
                float cg  = q_sigmoid(gc);
                float ng  = q_tanh(gni + rg * (accN[mt][r] + bnh));
                float hold = hc32[m * H_ + coln];
                float hy = ng + cg * (hold - ng);
                out[((size_t)t * B_ + m) * H_ + coln] = hy;
                hn32[m * H_ + coln] = hy;
                hn16[m * H_ + coln] = (_Float16)hy;
                if (t == T_ - 1)
                    out[(size_t)T_ * B_ * H_ + m * H_ + coln] = hy;   // `last`
            }
        }
        grid_barrier(bar, RBLOCKS * ++epoch);  // double-buffered h: 1 barrier/step
    }
}

// ---------------------------------------------------------------- launcher

extern "C" void kernel_launch(void* const* d_in, const int* in_sizes, int n_in,
                              void* d_out, int out_size, void* d_ws, size_t ws_size,
                              hipStream_t stream) {
    const float* x       = (const float*)d_in[0];
    const float* state   = (const float*)d_in[1];
    const float* w_ri    = (const float*)d_in[2];
    const float* w_ci    = (const float*)d_in[3];
    const float* w_ni    = (const float*)d_in[4];
    const float* w_rh    = (const float*)d_in[5];
    const float* w_ch    = (const float*)d_in[6];
    const float* w_nh    = (const float*)d_in[7];
    const float* bias_r  = (const float*)d_in[8];
    const float* bias_c  = (const float*)d_in[9];
    const float* bias_ni = (const float*)d_in[10];
    const float* bias_nh = (const float*)d_in[11];

    char* ws = (char*)d_ws;
    float*    scales = (float*)ws;                       // 6 floats
    unsigned* bar    = (unsigned*)(ws + 64);             // monotonic counter
    _Float16* WihT   = (_Float16*)(ws + 256);            // [NG, IN_]
    _Float16* WhhT   = WihT + (size_t)NG * IN_;          // [NG, H_]
    float*    hf32   = (float*)(WhhT + (size_t)NG * H_); // [2][B,H]
    _Float16* hf16   = (_Float16*)(hf32 + 2 * B_ * H_);  // [2][B,H]
    _Float16* pre    = hf16 + 2 * B_ * H_;               // [T*B, NG]

    (void)hipMemsetAsync(ws, 0, 256, stream);            // scales + barrier

    absmax_scale_kernel<<<6, 256, 0, stream>>>(w_ri, w_ci, w_ni,
                                               w_rh, w_ch, w_nh, scales);

    quant_transpose_kernel<<<(NG * IN_) / 256, 256, 0, stream>>>(
        w_ri, w_ci, w_ni, scales, 0, WihT);
    quant_transpose_kernel<<<(NG * H_) / 256, 256, 0, stream>>>(
        w_rh, w_ch, w_nh, scales, 3, WhhT);

    dim3 g3(M_ / 128, NG / 64);
    pre_gemm_kernel<<<g3, 256, 0, stream>>>(x, WihT, bias_r, bias_c, bias_ni, pre);

    const size_t smem_bytes = (size_t)(B_ + 96) * HPAD * sizeof(_Float16); // ~165 KB
    recurrent_kernel<<<RBLOCKS, RTPB, smem_bytes, stream>>>(
        pre, WhhT, state, bias_nh, (float*)d_out, hf32, hf16, bar);
}
